// MAB_50921132261692
// MI455X (gfx1250) — matrix operand order repaired and checked by reading, not compiled, not run on hardware
//
#include <hip/hip_runtime.h>
#include <hip/hip_bf16.h>
#include <cmath>

typedef __attribute__((ext_vector_type(16))) _Float16 v16h;
typedef __attribute__((ext_vector_type(8)))  _Float16 v8h;
typedef __attribute__((ext_vector_type(8)))  float    v8f;
typedef __attribute__((ext_vector_type(4)))  float    v4f;

#define BQ      4
#define NQS     2048
#define NKS     2048
#define DMODEL  512
#define NH      8
#define DHEAD   64

// ---------------------------------------------------------------------------
// WMMA helpers (gfx1250, wave32)
// ---------------------------------------------------------------------------
__device__ __forceinline__ v8f wmma_f16f32(v16h a, v16h b, v8f c) {
  // D = A(16x32 f16) * B(32x16 f16) + C(16x16 f32)
  return __builtin_amdgcn_wmma_f32_16x16x32_f16(false, a, false, b, (short)0, c,
                                                false, false);
}

// A-operand fragment: caller passes pointer already offset by (hi*16) halves.
// Halves [0..15] = K base..base+15 (contiguous).
__device__ __forceinline__ v16h fragA(const _Float16* p) {
  v8h a = *(const v8h*)p;
  v8h b = *(const v8h*)(p + 8);
  return __builtin_shufflevector(a, b, 0, 1, 2, 3, 4, 5, 6, 7,
                                 8, 9, 10, 11, 12, 13, 14, 15);
}

// B-operand fragment: 16 contiguous K values starting at K = hi*16
// (caller applies the hi*16 offset). One 32B contiguous load.
__device__ __forceinline__ v16h fragB(const _Float16* p) {
  return *(const v16h*)p;
}

__device__ __forceinline__ v8f zero8() {
  v8f z;
  for (int i = 0; i < 8; i++) z[i] = 0.f;
  return z;
}

// ---------------------------------------------------------------------------
// Elementwise cast / transpose prep kernels
// ---------------------------------------------------------------------------
__global__ void cast_f32_f16(const float* __restrict__ x,
                             _Float16* __restrict__ y, int n) {
  int i = blockIdx.x * blockDim.x + threadIdx.x;
  if (i < n) y[i] = (_Float16)x[i];
}

// W[k*N+n] (fan_in x fan_out, row = k) -> Wt[n*K+k] f16
__global__ void wtranspose_f16(const float* __restrict__ W,
                               _Float16* __restrict__ Wt, int K, int N) {
  int i = blockIdx.x * blockDim.x + threadIdx.x;
  if (i < K * N) {
    int k = i / N, n = i % N;
    Wt[(size_t)n * K + k] = (_Float16)W[i];
  }
}

// qkp[b,h,q] = <Qp[b,q,h*64:...], kp>
__global__ void qkp_dot(const float* __restrict__ Qp32,
                        const float* __restrict__ kp,
                        float* __restrict__ qkp) {
  int i = blockIdx.x * blockDim.x + threadIdx.x;
  if (i >= BQ * NH * NQS) return;
  int q = i % NQS;
  int h = (i / NQS) % NH;
  int b = i / (NQS * NH);
  const float* x = Qp32 + ((size_t)b * NQS + q) * DMODEL + h * DHEAD;
  float s = 0.f;
#pragma unroll
  for (int d = 0; d < DHEAD; d++) s += x[d] * kp[d];
  qkp[i] = s;
}

// ---------------------------------------------------------------------------
// Generic WMMA GEMM: C[M,N] = A[M,K](f16) * Bt[N,K]^T(f16) + bias
// Optional relu, optional f32 residual (added AFTER relu), outputs to any of
// Cf32 / Cf16 (scaled) / Ctv (V projection transposed to [b][h][d][key]).
// Block = 128 threads (4 waves), each wave owns a 16x64 tile of a 64x64 block.
// ---------------------------------------------------------------------------
__global__ __launch_bounds__(128)
void gemm_f16_wmma(const _Float16* __restrict__ A, const _Float16* __restrict__ Bt,
                   const float* __restrict__ bias, const float* __restrict__ resid,
                   float* __restrict__ Cf32, _Float16* __restrict__ Cf16,
                   _Float16* __restrict__ Ctv, float f16scale,
                   int M, int N, int K, int relu) {
  const int lane = threadIdx.x & 31;
  const int w = threadIdx.x >> 5;
  const int hi = lane >> 4;
  const int lm = lane & 15;
  const int m0 = blockIdx.y * 64 + w * 16;
  const int n0 = blockIdx.x * 64;

  v8f acc[4];
#pragma unroll
  for (int t = 0; t < 4; t++) acc[t] = zero8();

  const _Float16* arow = A + (size_t)(m0 + lm) * K + hi * 16;
  const _Float16* brow = Bt + (size_t)(n0 + lm) * K + hi * 16;

  for (int k0 = 0; k0 < K; k0 += 32) {
    v16h af = fragA(arow + k0);
#pragma unroll
    for (int t = 0; t < 4; t++) {
      acc[t] = wmma_f16f32(af, fragB(brow + (size_t)(t * 16) * K + k0), acc[t]);
    }
  }

#pragma unroll
  for (int t = 0; t < 4; t++) {
    int n = n0 + t * 16 + lm;
    float bv = bias ? bias[n] : 0.f;
#pragma unroll
    for (int r = 0; r < 8; r++) {
      int m = m0 + r + hi * 8;                 // C layout: lane=col, vgpr=row
      float v = acc[t][r] + bv;
      if (relu) v = fmaxf(v, 0.f);
      if (resid) v += resid[(size_t)m * N + n];
      if (Cf32) Cf32[(size_t)m * N + n] = v;
      if (Cf16) Cf16[(size_t)m * N + n] = (_Float16)(v * f16scale);
      if (Ctv) {                               // V: [b][h][d][key] transposed
        int b = m >> 11;                       // m / NKS
        int key = m & (NKS - 1);
        int h = n >> 6, d = n & 63;
        Ctv[((size_t)(b * NH + h) * DHEAD + d) * NKS + key] = (_Float16)v;
      }
    }
  }
}

// ---------------------------------------------------------------------------
// Fused flash attention with Shaw relative-position key/value biases.
// One wave handles 16 query rows of one (b,h); streams NK keys, 32 per step.
// S is computed TRANSPOSED (keys x queries) so each lane owns one query
// column -> online softmax is per-lane + one shfl_xor(16).
// Writes Opre = Qh + softmax(S) @ Vh + akp * vp   (pre-layernorm residual).
// ---------------------------------------------------------------------------
__global__ __launch_bounds__(256)
void attn_flash(const _Float16* __restrict__ Qs,    // [B*NQ][512] f16, pre-scaled
                const _Float16* __restrict__ Kh,    // [B*NK][512] f16
                const _Float16* __restrict__ Vt,    // [B][H][64][NK] f16
                const float* __restrict__ Kpos,     // [B][NQ][NK]
                const float* __restrict__ qkp,      // [B][H][NQ]
                const float* __restrict__ Qp32,     // [B*NQ][512] f32
                const float* __restrict__ vp,       // [64]
                float* __restrict__ Opre) {         // [B*NQ][512]
  const int lane = threadIdx.x & 31;
  const int w = threadIdx.x >> 5;
  const int hi = lane >> 4;
  const int lm = lane & 15;
  const int b = blockIdx.z, h = blockIdx.y;
  const int q = blockIdx.x * 128 + w * 16 + lm;    // this lane's query column

  const _Float16* qrow16 = Qs + ((size_t)b * NQS + q) * DMODEL + h * DHEAD;
  const v16h bq0 = fragB(qrow16 + hi * 16);        // B = Q^T, d = 0..31
  const v16h bq1 = fragB(qrow16 + 32 + hi * 16);   // d = 32..63
  const float qk = qkp[((size_t)b * NH + h) * NQS + q];
  const float* kposrow = Kpos + ((size_t)b * NQS + q) * NKS;
  const _Float16* kbase = Kh + (size_t)b * NKS * DMODEL + h * DHEAD + hi * 16;
  const _Float16* vbase = Vt + (size_t)(b * NH + h) * DHEAD * NKS + hi * 16;

  v8f o[4];
#pragma unroll
  for (int t = 0; t < 4; t++) o[t] = zero8();
  float mrow = -3.0e38f, lsum = 0.f, akp = 0.f;

  for (int k0 = 0; k0 < NKS; k0 += 32) {
    // S^T tile: 32 keys x 16 queries, A = K-tile, B = Q^T
    const _Float16* kp0 = kbase + (size_t)(k0 + lm) * DMODEL;
    v8f s0 = wmma_f16f32(fragA(kp0), bq0, zero8());
    s0 = wmma_f16f32(fragA(kp0 + 32), bq1, s0);
    const _Float16* kp1 = kbase + (size_t)(k0 + 16 + lm) * DMODEL;
    v8f s1 = wmma_f16f32(fragA(kp1), bq0, zero8());
    s1 = wmma_f16f32(fragA(kp1 + 32), bq1, s1);

    // Shaw key bias: key for acc element r is k0 (+16) + hi*8 + r
    v4f ka = *(const v4f*)(kposrow + k0 + hi * 8);
    v4f kb = *(const v4f*)(kposrow + k0 + hi * 8 + 4);
    v4f kc = *(const v4f*)(kposrow + k0 + 16 + hi * 8);
    v4f kd = *(const v4f*)(kposrow + k0 + 16 + hi * 8 + 4);
    float kv0[8], kv1[8];
#pragma unroll
    for (int r = 0; r < 4; r++) {
      kv0[r] = ka[r]; kv0[4 + r] = kb[r];
      kv1[r] = kc[r]; kv1[4 + r] = kd[r];
    }
    float tmax = -3.0e38f;
#pragma unroll
    for (int r = 0; r < 8; r++) {
      s0[r] += kv0[r] * qk;
      s1[r] += kv1[r] * qk;
      tmax = fmaxf(tmax, fmaxf(s0[r], s1[r]));
    }
    tmax = fmaxf(tmax, __shfl_xor(tmax, 16, 32));  // combine key halves of q
    const float mnew = fmaxf(mrow, tmax);
    const float alpha = __expf(mrow - mnew);
    lsum *= alpha; akp *= alpha;
#pragma unroll
    for (int t = 0; t < 4; t++)
#pragma unroll
      for (int r = 0; r < 8; r++) o[t][r] *= alpha;

    float p0[8], p1[8];
#pragma unroll
    for (int r = 0; r < 8; r++) {
      p0[r] = __expf(s0[r] - mnew);
      p1[r] = __expf(s1[r] - mnew);
      lsum += p0[r] + p1[r];
      akp += p0[r] * kv0[r] + p1[r] * kv1[r];     // Shaw value bias accumulator
    }
    mrow = mnew;

    // Pack P into B-operand layout: lane needs keys hi*16 + [0..16); the
    // missing key-half lives in the partner lane (lane^16).
    v16h pf;
#pragma unroll
    for (int r = 0; r < 8; r++) {
      float x0 = __shfl_xor(p0[r], 16, 32);
      float x1 = __shfl_xor(p1[r], 16, 32);
      pf[r]     = (_Float16)(hi ? x1 : p0[r]);    // keys hi*16 + r
      pf[r + 8] = (_Float16)(hi ? p1[r] : x0);    // keys hi*16 + 8 + r
    }

    // out^T += V^T(16d x 32keys) * P(32keys x 16q)
#pragma unroll
    for (int t = 0; t < 4; t++) {
      const _Float16* vptr = vbase + (size_t)(t * 16 + lm) * NKS + k0;
      o[t] = wmma_f16f32(fragA(vptr), pf, o[t]);
    }
  }

  const float ltot = lsum + __shfl_xor(lsum, 16, 32);
  const float atot = akp + __shfl_xor(akp, 16, 32);
  const float inv = 1.f / ltot;
  float* orow = Opre + ((size_t)b * NQS + q) * DMODEL + h * DHEAD;
  const float* q32 = Qp32 + ((size_t)b * NQS + q) * DMODEL + h * DHEAD;
#pragma unroll
  for (int t = 0; t < 4; t++)
#pragma unroll
    for (int r = 0; r < 8; r++) {
      int d = t * 16 + hi * 8 + r;                // contiguous in r -> 32B store
      orow[d] = o[t][r] * inv + q32[d] + atot * inv * vp[d];
    }
}

// ---------------------------------------------------------------------------
// LayerNorm: one wave per 512-wide row; wave-only shfl reduction.
// ---------------------------------------------------------------------------
__global__ __launch_bounds__(256)
void layernorm_rows(const float* __restrict__ X, const float* __restrict__ g,
                    const float* __restrict__ be, float* __restrict__ Y32,
                    _Float16* __restrict__ Y16, int Mrows) {
  const int lane = threadIdx.x & 31;
  const int w = threadIdx.x >> 5;
  const int row = blockIdx.x * 8 + w;
  if (row >= Mrows) return;
  const float* x = X + (size_t)row * DMODEL;
  float vals[16], s = 0.f, s2 = 0.f;
#pragma unroll
  for (int i = 0; i < 16; i++) {
    float v = x[lane + i * 32];
    vals[i] = v; s += v; s2 += v * v;
  }
#pragma unroll
  for (int off = 16; off >= 1; off >>= 1) {
    s += __shfl_xor(s, off, 32);
    s2 += __shfl_xor(s2, off, 32);
  }
  const float mean = s * (1.f / DMODEL);
  const float var = s2 * (1.f / DMODEL) - mean * mean;
  const float rinv = rsqrtf(var + 1e-5f);
#pragma unroll
  for (int i = 0; i < 16; i++) {
    int c = lane + i * 32;
    float y = (vals[i] - mean) * rinv * g[c] + be[c];
    if (Y32) Y32[(size_t)row * DMODEL + c] = y;
    if (Y16) Y16[(size_t)row * DMODEL + c] = (_Float16)y;
  }
}

// ---------------------------------------------------------------------------
extern "C" void kernel_launch(void* const* d_in, const int* in_sizes, int n_in,
                              void* d_out, int out_size, void* d_ws, size_t ws_size,
                              hipStream_t stream) {
  (void)in_sizes; (void)n_in; (void)out_size; (void)ws_size;
  const float* Q    = (const float*)d_in[0];
  const float* K    = (const float*)d_in[1];
  const float* Kpos = (const float*)d_in[2];
  const float* Wq   = (const float*)d_in[3];
  const float* bq   = (const float*)d_in[4];
  const float* Wk   = (const float*)d_in[5];
  const float* bk   = (const float*)d_in[6];
  const float* Wv   = (const float*)d_in[7];
  const float* bv   = (const float*)d_in[8];
  const float* kp   = (const float*)d_in[9];
  const float* vp   = (const float*)d_in[10];
  const float* W0   = (const float*)d_in[11];
  const float* b0   = (const float*)d_in[12];
  const float* W1   = (const float*)d_in[13];
  const float* b1   = (const float*)d_in[14];
  const float* g0   = (const float*)d_in[15];
  const float* be0  = (const float*)d_in[16];
  const float* g1   = (const float*)d_in[17];
  const float* be1  = (const float*)d_in[18];
  float* out = (float*)d_out;

  char* ws = (char*)d_ws;
  size_t off = 0;
  auto alloc = [&](size_t bytes) -> char* {
    char* p = ws + off;
    off += (bytes + 255) & ~(size_t)255;
    return p;
  };
  const size_t Mrows = (size_t)BQ * NQS;                 // 8192
  const size_t NE = Mrows * DMODEL;                      // 4,194,304
  _Float16* Qc   = (_Float16*)alloc(NE * 2);
  _Float16* Kc   = (_Float16*)alloc(NE * 2);
  _Float16* Wqt  = (_Float16*)alloc((size_t)DMODEL * DMODEL * 2);
  _Float16* Wkt  = (_Float16*)alloc((size_t)DMODEL * DMODEL * 2);
  _Float16* Wvt  = (_Float16*)alloc((size_t)DMODEL * DMODEL * 2);
  _Float16* W0t  = (_Float16*)alloc((size_t)DMODEL * DMODEL * 2);
  _Float16* W1t  = (_Float16*)alloc((size_t)DMODEL * DMODEL * 2);
  float*    Qp32 = (float*)   alloc(NE * 4);
  _Float16* Qsc  = (_Float16*)alloc(NE * 2);             // Qp * 1/sqrt(512), f16
  _Float16* Kp16 = (_Float16*)alloc(NE * 2);
  _Float16* Vt   = (_Float16*)alloc(NE * 2);             // [b][h][d][key]
  float*    qkpv = (float*)   alloc((size_t)BQ * NH * NQS * 4);
  float*    Opre = (float*)   alloc(NE * 4);
  float*    Xn32 = (float*)   alloc(NE * 4);
  _Float16* Xn16 = (_Float16*)alloc(NE * 2);
  _Float16* H1   = (_Float16*)alloc(NE * 2);
  float*    O2   = (float*)   alloc(NE * 4);

  const int nel = (int)NE;
  cast_f32_f16<<<(nel + 255) / 256, 256, 0, stream>>>(Q, Qc, nel);
  cast_f32_f16<<<(nel + 255) / 256, 256, 0, stream>>>(K, Kc, nel);
  const int wn = DMODEL * DMODEL;
  wtranspose_f16<<<(wn + 255) / 256, 256, 0, stream>>>(Wq, Wqt, DMODEL, DMODEL);
  wtranspose_f16<<<(wn + 255) / 256, 256, 0, stream>>>(Wk, Wkt, DMODEL, DMODEL);
  wtranspose_f16<<<(wn + 255) / 256, 256, 0, stream>>>(Wv, Wvt, DMODEL, DMODEL);
  wtranspose_f16<<<(wn + 255) / 256, 256, 0, stream>>>(W0, W0t, DMODEL, DMODEL);
  wtranspose_f16<<<(wn + 255) / 256, 256, 0, stream>>>(W1, W1t, DMODEL, DMODEL);

  dim3 gg(DMODEL / 64, (unsigned)(Mrows / 64));          // (8,128) x 128thr
  const float scl = 1.0f / sqrtf((float)DMODEL);
  gemm_f16_wmma<<<gg, 128, 0, stream>>>(Qc, Wqt, bq, nullptr, Qp32, Qsc, nullptr,
                                        scl, (int)Mrows, DMODEL, DMODEL, 0);
  gemm_f16_wmma<<<gg, 128, 0, stream>>>(Kc, Wkt, bk, nullptr, nullptr, Kp16, nullptr,
                                        1.f, (int)Mrows, DMODEL, DMODEL, 0);
  gemm_f16_wmma<<<gg, 128, 0, stream>>>(Kc, Wvt, bv, nullptr, nullptr, nullptr, Vt,
                                        1.f, (int)Mrows, DMODEL, DMODEL, 0);

  const int nqk = BQ * NH * NQS;
  qkp_dot<<<(nqk + 255) / 256, 256, 0, stream>>>(Qp32, kp, qkpv);

  dim3 ag(NQS / 128, NH, BQ);
  attn_flash<<<ag, 256, 0, stream>>>(Qsc, Kp16, Vt, Kpos, qkpv, Qp32, vp, Opre);

  layernorm_rows<<<(unsigned)(Mrows / 8), 256, 0, stream>>>(Opre, g0, be0, Xn32,
                                                            Xn16, (int)Mrows);
  gemm_f16_wmma<<<gg, 128, 0, stream>>>(Xn16, W0t, b0, nullptr, nullptr, H1, nullptr,
                                        1.f, (int)Mrows, DMODEL, DMODEL, 1);
  gemm_f16_wmma<<<gg, 128, 0, stream>>>(H1, W1t, b1, Xn32, O2, nullptr, nullptr,
                                        1.f, (int)Mrows, DMODEL, DMODEL, 1);
  layernorm_rows<<<(unsigned)(Mrows / 8), 256, 0, stream>>>(O2, g1, be1, out,
                                                            nullptr, (int)Mrows);
}
